// AttenConv1d_74002286510474
// MI455X (gfx1250) — matrix operand order repaired and checked by reading, not compiled, not run on hardware
//
#include <hip/hip_runtime.h>
#include <math.h>

typedef __attribute__((ext_vector_type(16))) _Float16 v16h;
typedef __attribute__((ext_vector_type(8)))  _Float16 v8h;
typedef __attribute__((ext_vector_type(8)))  float    v8f;

constexpr int BN   = 16384;  // nodes per batch
constexpr int KNB  = 16;     // neighbors per node
constexpr int C    = 128;    // channels (C_IN == C_OUT)
constexpr int NODES_PER_BLOCK = 16;
constexpr int THREADS = 512; // 16 waves (wave32)

// Pack 8 floats at p0 and 8 floats at p1 into a v16h (f32 -> f16).
__device__ inline v16h pack16(const float* __restrict__ p0,
                              const float* __restrict__ p1) {
    float4 f0 = ((const float4*)p0)[0];
    float4 f1 = ((const float4*)p0)[1];
    float4 f2 = ((const float4*)p1)[0];
    float4 f3 = ((const float4*)p1)[1];
    v16h a;
    a[0]  = (_Float16)f0.x; a[1]  = (_Float16)f0.y; a[2]  = (_Float16)f0.z; a[3]  = (_Float16)f0.w;
    a[4]  = (_Float16)f1.x; a[5]  = (_Float16)f1.y; a[6]  = (_Float16)f1.z; a[7]  = (_Float16)f1.w;
    a[8]  = (_Float16)f2.x; a[9]  = (_Float16)f2.y; a[10] = (_Float16)f2.z; a[11] = (_Float16)f2.w;
    a[12] = (_Float16)f3.x; a[13] = (_Float16)f3.y; a[14] = (_Float16)f3.z; a[15] = (_Float16)f3.w;
    return a;
}

__global__ __launch_bounds__(THREADS)
void attenconv_wmma_kernel(const float* __restrict__ x,
                           const int*   __restrict__ eidx,   // [2,B,N,K]
                           const float* __restrict__ Wm,     // [C_OUT, C_IN]
                           const float* __restrict__ bias,   // [C_OUT]
                           float*       __restrict__ out,    // [B,N,C_OUT]
                           int nTotal)                       // B*BN
{
    __shared__ __align__(16) _Float16 s_v[NODES_PER_BLOCK][C]; // (x + h), f16
    __shared__ float s_q[NODES_PER_BLOCK][KNB];                // colsum of p_attn

    const int tid  = threadIdx.x;
    const int wave = tid >> 5;
    const int lane = tid & 31;
    const int hf   = lane >> 4;   // 0: lanes 0-15, 1: lanes 16-31
    const int l16  = lane & 15;

    const int Bb = nTotal / BN;                     // batch count
    const int g  = blockIdx.x * NODES_PER_BLOCK + wave;  // global node id
    const int b  = g / BN;
    const int n  = g % BN;

    const float* xb = x + (size_t)b * BN * C;
    const size_t e0_base = ((size_t)b * BN + n) * KNB;            // edge_index[0] (neighbors, x_j)
    const size_t e1_base = (((size_t)Bb + b) * BN + n) * KNB;     // edge_index[1] (centers,  x_i)

    // ---- Phase 1: scores = Xi @ Xj^T via WMMA (16x16 output, K=128 in 4 chunks)
    const int   rowSel   = l16;                       // this lane's M (A) / N (B) index
    const int   iCenter  = eidx[e1_base + rowSel];    // A-matrix row source (x_i)
    const int   iNeigh   = eidx[e0_base + rowSel];    // B-matrix column source (x_j row)
    const float* rowA = xb + (size_t)iCenter * C;
    const float* rowB = xb + (size_t)iNeigh  * C;

    v8f cs = {};
    #pragma unroll
    for (int kk = 0; kk < 4; ++kk) {
        // A (x_i 16x32 chunk): halves 0..7 = K a0..a0+7, halves 8..15 = K a0+16..a0+23
        const int a0 = kk * 32 + hf * 8;
        v16h af = pack16(rowA + a0, rowA + a0 + 16);
        // B (x_j^T 32x16 chunk): halves 0..7 = K b0..b0+7, halves 8..15 = K b0+16..b0+23
        const int b0 = kk * 32 + hf * 8;
        v16h bf = pack16(rowB + b0, rowB + b0 + 16);
        cs = __builtin_amdgcn_wmma_f32_16x16x32_f16(false, af, false, bf,
                                                    (short)0, cs, false, false);
    }

    // ---- Phase 2: softmax over j (the N dimension -> across each 16-lane half)
    const float scale = 0.08838834764831845f;   // 1/sqrt(128)
    float p[8];
    #pragma unroll
    for (int v = 0; v < 8; ++v) {
        float s = cs[v] * scale;
        float m = s;
        m = fmaxf(m, __shfl_xor(m, 1));
        m = fmaxf(m, __shfl_xor(m, 2));
        m = fmaxf(m, __shfl_xor(m, 4));
        m = fmaxf(m, __shfl_xor(m, 8));
        float e = __expf(s - m);
        float t = e;
        t += __shfl_xor(t, 1);
        t += __shfl_xor(t, 2);
        t += __shfl_xor(t, 4);
        t += __shfl_xor(t, 8);
        p[v] = e / t;
    }
    // q[j] = sum_k p_attn[k][j] : per-lane sum over its 8 rows, then combine halves
    float q = p[0] + p[1] + p[2] + p[3] + p[4] + p[5] + p[6] + p[7];
    q += __shfl_xor(q, 16);
    if (hf == 0) s_q[wave][l16] = q;
    __syncthreads();

    // ---- Phase 3: v = x_center + sum_j q[j] * x_j[j][:]  (each lane: 4 channels)
    {
        const int cch = lane * 4;
        float ax = 0.f, ay = 0.f, az = 0.f, aw = 0.f;
        #pragma unroll
        for (int j = 0; j < KNB; ++j) {
            int ij = eidx[e0_base + j];
            float4 xv = *(const float4*)(xb + (size_t)ij * C + cch);
            float qj = s_q[wave][j];
            ax = fmaf(qj, xv.x, ax);
            ay = fmaf(qj, xv.y, ay);
            az = fmaf(qj, xv.z, az);
            aw = fmaf(qj, xv.w, aw);
        }
        float4 xc = *(const float4*)(xb + (size_t)n * C + cch);
        s_v[wave][cch + 0] = (_Float16)(xc.x + ax);
        s_v[wave][cch + 1] = (_Float16)(xc.y + ay);
        s_v[wave][cch + 2] = (_Float16)(xc.z + az);
        s_v[wave][cch + 3] = (_Float16)(xc.w + aw);
    }
    __syncthreads();

    // ---- Phase 4: batched linear  Y(16x128) = V(16x128) @ W^T(128x128), relu
    // Waves 0..7 each own one 16-wide c_out tile; 4 K-chunks of WMMA each.
    if (wave < 8) {
        const int co = wave * 16 + l16;              // this lane's output channel (N)
        const float* wrow = Wm + (size_t)co * C;     // B[k][n] = W[n][k] -> row of W
        v8f acc = {};
        #pragma unroll
        for (int kk = 0; kk < 4; ++kk) {
            const int a0 = kk * 32 + hf * 8;
            v8h va0 = *(const v8h*)&s_v[l16][a0];
            v8h va1 = *(const v8h*)&s_v[l16][a0 + 16];
            v16h af;
            #pragma unroll
            for (int i = 0; i < 8; ++i) { af[i] = va0[i]; af[8 + i] = va1[i]; }
            const int b0 = kk * 32 + hf * 8;
            v16h bf = pack16(wrow + b0, wrow + b0 + 16);
            acc = __builtin_amdgcn_wmma_f32_16x16x32_f16(false, af, false, bf,
                                                         (short)0, acc, false, false);
        }
        const float bv = bias[co];
        #pragma unroll
        for (int v = 0; v < 8; ++v) {
            const int node = blockIdx.x * NODES_PER_BLOCK + v + hf * 8;  // C-frag row M
            float r = acc[v] + bv;
            out[(size_t)node * C + co] = fmaxf(r, 0.0f);
        }
    }
}

extern "C" void kernel_launch(void* const* d_in, const int* in_sizes, int n_in,
                              void* d_out, int out_size, void* d_ws, size_t ws_size,
                              hipStream_t stream) {
    const float* x    = (const float*)d_in[0];
    const int*   eidx = (const int*)d_in[1];
    const float* Wm   = (const float*)d_in[2];
    const float* bias = (const float*)d_in[3];
    float*       out  = (float*)d_out;

    const int nTotal = in_sizes[0] / C;            // B * N nodes
    const int blocks = nTotal / NODES_PER_BLOCK;   // 2048 for B=2,N=16384

    hipLaunchKernelGGL(attenconv_wmma_kernel, dim3(blocks), dim3(THREADS), 0, stream,
                       x, eidx, Wm, bias, out, nTotal);
}
